// MultiHeadAttention_14499809591541
// MI455X (gfx1250) — compile-verified, matrix operand order checked by reading, not run on hardware
//
#include <hip/hip_runtime.h>

#define D_MODEL 1024
#define NHEADS  16
#define HDIM    64
#define BATCH   4
#define SEQ     2048

typedef __attribute__((ext_vector_type(16))) __bf16 v16bf;
typedef __attribute__((ext_vector_type(8)))  __bf16 v8bf;
typedef __attribute__((ext_vector_type(8)))  float  v8f;

static __device__ __forceinline__ __bf16 f2bf(float f) {
    union { float f; unsigned u; } v; v.f = f;
    unsigned r = v.u + 0x7FFFu + ((v.u >> 16) & 1u);   // round-to-nearest-even
    unsigned short h = (unsigned short)(r >> 16);
    return __builtin_bit_cast(__bf16, h);
}

// Wave-relative 32-bit LDS byte address (low 32 bits of the generic pointer,
// per the flat-aperture mapping).
static __device__ __forceinline__ unsigned lds_addr(const void* p) {
    return (unsigned)(unsigned long long)p;
}

// Async DMA: global memory -> LDS, 16 bytes per lane, tracked by ASYNCcnt.
static __device__ __forceinline__ void async_copy_b128(unsigned lds_off, const void* gaddr) {
    asm volatile("global_load_async_to_lds_b128 %0, %1, off"
                 :: "v"(lds_off), "v"(gaddr) : "memory");
}
static __device__ __forceinline__ void wait_async0() {
    asm volatile("s_wait_asynccnt 0x0" ::: "memory");
}

// Load one 16x32 bf16 WMMA A/B fragment from a row-major [rows][stride] array.
// A-matrix: lane&15 = M row, kb = (lane>>4)*8 selects K window per ISA layout:
//   frag[0..7]  = K = kb .. kb+7
//   frag[8..15] = K = kb+16 .. kb+23
// B-matrix uses the mirrored layout (lane&15 = N column), so storing B columns
// as rows (i.e. [n][k]) lets the same loader serve both.
static __device__ __forceinline__ v16bf load_frag(const __bf16* base, int row0, int stride, int lane) {
    const int r  = lane & 15;
    const int kb = (lane >> 4) << 3;
    const __bf16* p = base + (size_t)(row0 + r) * stride + kb;
    v8bf lo = *(const v8bf*)p;
    v8bf hi = *(const v8bf*)(p + 16);
    return __builtin_shufflevector(lo, hi, 0, 1, 2, 3, 4, 5, 6, 7,
                                           8, 9, 10, 11, 12, 13, 14, 15);
}

// ---------------------------------------------------------------------------
// 1) f32 -> bf16 weight conversion
// ---------------------------------------------------------------------------
__global__ __launch_bounds__(256) void k_f32_to_bf16(const float* __restrict__ src,
                                                     __bf16* __restrict__ dst, int n) {
    int idx = (blockIdx.x * 256 + threadIdx.x) * 4;
    if (idx + 3 < n) {
        float4 v = *(const float4*)(src + idx);
        dst[idx + 0] = f2bf(v.x);
        dst[idx + 1] = f2bf(v.y);
        dst[idx + 2] = f2bf(v.z);
        dst[idx + 3] = f2bf(v.w);
    }
}

// ---------------------------------------------------------------------------
// 2) Q/K/V projection: out(i,j) = sum_k x[i,k] * W[j,k]   (x @ W^T)
//    blockIdx.z: 0=Q, 1=K (layout [B,H,S,Dh]), 2=V (stored transposed [B,H,Dh,S])
//    Tile: 128(M) x 64(N), BK=32, double-buffered LDS, one barrier/iteration.
//    W tile: async DMA.  X tile: f32 loads issued BEFORE the compute of the
//    current tile (latency hidden under WMMA), converted+stored after it.
// ---------------------------------------------------------------------------
__global__ __launch_bounds__(256) void k_proj(const float* __restrict__ xq,
                                              const float* __restrict__ xk,
                                              const float* __restrict__ xv,
                                              const __bf16* __restrict__ wall,
                                              __bf16* __restrict__ outq,
                                              __bf16* __restrict__ outk,
                                              __bf16* __restrict__ outv) {
    const int z = blockIdx.z;
    const float*  x   = (z == 0) ? xq : (z == 1) ? xk : xv;
    const __bf16* W   = wall + (size_t)z * D_MODEL * D_MODEL;
    __bf16*       out = (z == 0) ? outq : (z == 1) ? outk : outv;

    __shared__ __bf16 Xs[2][128 * 40];   // 32 K + 8 pad (16B aligned, bank skew)
    __shared__ __bf16 Ws[2][64 * 40];

    const int t = threadIdx.x;
    const int lane = t & 31, wid = t >> 5;
    const int wm = wid & 3, wn = wid >> 2;           // 4x2 wave grid
    const int M0 = blockIdx.y * 128;
    const int N0 = blockIdx.x * 64;

    // staging geometry
    const int wrow = t >> 2;                 // 0..63
    const int wcc  = (t & 3) * 8;            // 0,8,16,24
    const int xrr  = t >> 3;                 // 0..31
    const int xcc  = (t & 7) * 4;            // 0..28

    v8f acc[4] = {};

    // ---- prologue: stage tile k0=0 into buffer 0 ----
    async_copy_b128(lds_addr(&Ws[0][wrow * 40 + wcc]),
                    W + (size_t)(N0 + wrow) * D_MODEL + wcc);
    {
#pragma unroll
        for (int i = 0; i < 4; ++i) {
            const int row = xrr + i * 32;
            float4 v = *(const float4*)(x + (size_t)(M0 + row) * D_MODEL + xcc);
            __bf16* d = &Xs[0][row * 40 + xcc];
            d[0] = f2bf(v.x); d[1] = f2bf(v.y); d[2] = f2bf(v.z); d[3] = f2bf(v.w);
        }
    }

    for (int k0 = 0; k0 < D_MODEL; k0 += 32) {
        const int cur = (k0 >> 5) & 1;
        const int nxt = cur ^ 1;
        const bool more = (k0 + 32) < D_MODEL;

        wait_async0();          // my async loads for buffer `cur` are done
        __syncthreads();        // everyone's loads done + everyone done reading `nxt`

        // issue next tile's W DMA and X f32 loads now; they overlap the WMMAs
        float4 xn[4];
        if (more) {
            async_copy_b128(lds_addr(&Ws[nxt][wrow * 40 + wcc]),
                            W + (size_t)(N0 + wrow) * D_MODEL + (k0 + 32) + wcc);
#pragma unroll
            for (int i = 0; i < 4; ++i)
                xn[i] = *(const float4*)(x + (size_t)(M0 + xrr + i * 32) * D_MODEL
                                           + (k0 + 32) + xcc);
        }

        v16bf a0 = load_frag(Xs[cur], wm * 32,      40, lane);
        v16bf a1 = load_frag(Xs[cur], wm * 32 + 16, 40, lane);
        v16bf b0 = load_frag(Ws[cur], wn * 32,      40, lane);
        v16bf b1 = load_frag(Ws[cur], wn * 32 + 16, 40, lane);
        acc[0] = __builtin_amdgcn_wmma_f32_16x16x32_bf16(false, a0, false, b0, (short)0, acc[0], false, false);
        acc[1] = __builtin_amdgcn_wmma_f32_16x16x32_bf16(false, a0, false, b1, (short)0, acc[1], false, false);
        acc[2] = __builtin_amdgcn_wmma_f32_16x16x32_bf16(false, a1, false, b0, (short)0, acc[2], false, false);
        acc[3] = __builtin_amdgcn_wmma_f32_16x16x32_bf16(false, a1, false, b1, (short)0, acc[3], false, false);

        if (more) {
#pragma unroll
            for (int i = 0; i < 4; ++i) {
                __bf16* d = &Xs[nxt][(xrr + i * 32) * 40 + xcc];
                d[0] = f2bf(xn[i].x); d[1] = f2bf(xn[i].y);
                d[2] = f2bf(xn[i].z); d[3] = f2bf(xn[i].w);
            }
        }
    }

    // C layout: lane&15 = N, VGPR r -> M = r + 8*(lane>>4)
    const int hf = lane >> 4;
    const int nl = lane & 15;
#pragma unroll
    for (int mi = 0; mi < 2; ++mi)
#pragma unroll
    for (int ni = 0; ni < 2; ++ni)
#pragma unroll
    for (int r = 0; r < 8; ++r) {
        const int g = M0 + wm * 32 + mi * 16 + r + 8 * hf;   // row in [0, B*S)
        const int c = N0 + wn * 32 + ni * 16 + nl;           // col in [0, D)
        const int b = g >> 11;        // / SEQ
        const int s = g & (SEQ - 1);
        const int h = c >> 6;         // / HDIM
        const int d = c & (HDIM - 1);
        const size_t idx = (z == 2)
            ? (((size_t)(b * NHEADS + h) * HDIM + d) * SEQ + s)   // V^T: [B,H,Dh,S]
            : (((size_t)(b * NHEADS + h) * SEQ + s) * HDIM + d);  // Q,K: [B,H,S,Dh]
        out[idx] = f2bf(acc[mi * 2 + ni][r]);
    }
}

// ---------------------------------------------------------------------------
// 3) Causal flash attention. Block = 64 queries (4 waves x 16 rows), key
//    blocks of 64 staged via async DMA, double-buffered (one barrier/iter).
//    Online softmax; P converted C-layout->A-layout through a per-wave LDS
//    tile (LDS is in-order per wave, no barrier needed for that region).
// ---------------------------------------------------------------------------
__global__ __launch_bounds__(128) void k_flash(const __bf16* __restrict__ Q,
                                               const __bf16* __restrict__ K,
                                               const __bf16* __restrict__ Vt,
                                               __bf16* __restrict__ O) {
    __shared__ __bf16 Ks[2][64 * 72];     // [key][d]
    __shared__ __bf16 Vs[2][64 * 72];     // [d][key]  (from V^T)
    __shared__ __bf16 Ps[4 * 16 * 72];    // per-wave P staging [16 q rows][64 keys]

    const int t = threadIdx.x, lane = t & 31, wid = t >> 5;
    const int bh = blockIdx.y;
    const int b = bh >> 4, h = bh & (NHEADS - 1);
    const int q0 = blockIdx.x * 64;
    const int jmax = (int)blockIdx.x;

    const __bf16* Qh = Q  + (size_t)bh * SEQ * HDIM;
    const __bf16* Kh = K  + (size_t)bh * SEQ * HDIM;
    const __bf16* Vh = Vt + (size_t)bh * HDIM * SEQ;

    const int hf = lane >> 4, nl = lane & 15, kb = hf * 8;
    const int qrow = q0 + wid * 16 + nl;

    // Q fragments held in registers for the whole kernel (Dh=64 -> 2 K-windows)
    v16bf aq[2];
#pragma unroll
    for (int w = 0; w < 2; ++w) {
        const __bf16* p = Qh + (size_t)qrow * HDIM + w * 32 + kb;
        v8bf lo = *(const v8bf*)p;
        v8bf hi = *(const v8bf*)(p + 16);
        aq[w] = __builtin_shufflevector(lo, hi, 0, 1, 2, 3, 4, 5, 6, 7,
                                                8, 9, 10, 11, 12, 13, 14, 15);
    }

    float m[8], l[8];
    v8f o[4] = {};
#pragma unroll
    for (int r = 0; r < 8; ++r) { m[r] = -3.0e38f; l[r] = 0.0f; }

    __bf16* Psw = &Ps[wid * 16 * 72];

    // staging geometry (per thread): one row, half of its 64 columns
    const int srow = t >> 1;
    const int scb  = (t & 1) * 32;

    // ---- prologue: stage key block 0 into buffer 0 ----
    {
        const __bf16* sk = Kh + (size_t)srow * HDIM + scb;
        const __bf16* sv = Vh + (size_t)srow * SEQ + scb;
        const unsigned kd = lds_addr(&Ks[0][srow * 72 + scb]);
        const unsigned vd = lds_addr(&Vs[0][srow * 72 + scb]);
#pragma unroll
        for (int i = 0; i < 4; ++i) {
            async_copy_b128(kd + i * 16, sk + i * 8);
            async_copy_b128(vd + i * 16, sv + i * 8);
        }
    }

    for (int j = 0; j <= jmax; ++j) {
        const int k0  = j * 64;
        const int cur = j & 1;

        wait_async0();          // my DMA into buffer `cur` is done
        __syncthreads();        // everyone's DMA done + everyone done reading cur^1

        // issue next key block's DMA; overlaps all compute below
        if (j < jmax) {
            const int nxt = cur ^ 1;
            const __bf16* sk = Kh + (size_t)(k0 + 64 + srow) * HDIM + scb;
            const __bf16* sv = Vh + (size_t)srow * SEQ + (k0 + 64) + scb;
            const unsigned kd = lds_addr(&Ks[nxt][srow * 72 + scb]);
            const unsigned vd = lds_addr(&Vs[nxt][srow * 72 + scb]);
#pragma unroll
            for (int i = 0; i < 4; ++i) {
                async_copy_b128(kd + i * 16, sk + i * 8);
                async_copy_b128(vd + i * 16, sv + i * 8);
            }
        }

        // scores S = Q K^T / sqrt(Dh): 4 N-tiles of keys x 2 K-windows of d
        v8f sa[4] = {};
#pragma unroll
        for (int n = 0; n < 4; ++n) {
#pragma unroll
            for (int dw = 0; dw < 2; ++dw) {
                v16bf bk = load_frag(Ks[cur] + dw * 32, n * 16, 72, lane);
                sa[n] = __builtin_amdgcn_wmma_f32_16x16x32_bf16(false, aq[dw], false, bk,
                                                                (short)0, sa[n], false, false);
            }
        }

        // scale + causal mask (only the diagonal key block needs masking)
        const bool diag = (j == jmax);
#pragma unroll
        for (int n = 0; n < 4; ++n)
#pragma unroll
        for (int r = 0; r < 8; ++r) {
            float s = sa[n][r] * 0.125f;                 // 1/sqrt(64)
            if (diag) {
                const int kg = k0 + n * 16 + nl;
                const int qg = q0 + wid * 16 + r + 8 * hf;
                if (kg > qg) s = -1.0e9f;
            }
            sa[n][r] = s;
        }

        // online softmax: row max/sum across the 16 lanes of each half-wave
#pragma unroll
        for (int r = 0; r < 8; ++r) {
            float mr = -3.0e38f;
#pragma unroll
            for (int n = 0; n < 4; ++n) mr = fmaxf(mr, sa[n][r]);
#pragma unroll
            for (int off = 8; off >= 1; off >>= 1)
                mr = fmaxf(mr, __shfl_xor(mr, off, 32));
            const float mnew  = fmaxf(m[r], mr);
            const float scale = __expf(m[r] - mnew);
            float ls = 0.0f;
#pragma unroll
            for (int n = 0; n < 4; ++n) {
                float p = __expf(sa[n][r] - mnew);
                sa[n][r] = p;
                ls += p;
            }
#pragma unroll
            for (int off = 8; off >= 1; off >>= 1)
                ls += __shfl_xor(ls, off, 32);
            l[r] = l[r] * scale + ls;
            m[r] = mnew;
#pragma unroll
            for (int n = 0; n < 4; ++n) o[n][r] *= scale;
        }

        // C-layout -> A-layout via per-wave LDS tile (in-order per wave)
#pragma unroll
        for (int n = 0; n < 4; ++n)
#pragma unroll
        for (int r = 0; r < 8; ++r)
            Psw[(r + 8 * hf) * 72 + n * 16 + nl] = f2bf(sa[n][r]);

        // O += P V : A = P (16x64 keys), B = V (keys x d) from Vs=[d][key]
        v16bf ap0 = load_frag(Psw,      0, 72, lane);
        v16bf ap1 = load_frag(Psw + 32, 0, 72, lane);
#pragma unroll
        for (int nd = 0; nd < 4; ++nd) {
            v16bf bv0 = load_frag(Vs[cur],      nd * 16, 72, lane);
            v16bf bv1 = load_frag(Vs[cur] + 32, nd * 16, 72, lane);
            o[nd] = __builtin_amdgcn_wmma_f32_16x16x32_bf16(false, ap0, false, bv0,
                                                            (short)0, o[nd], false, false);
            o[nd] = __builtin_amdgcn_wmma_f32_16x16x32_bf16(false, ap1, false, bv1,
                                                            (short)0, o[nd], false, false);
        }
    }

    // normalize and store to [B,S,D] bf16
#pragma unroll
    for (int r = 0; r < 8; ++r) {
        const float inv = 1.0f / l[r];
        const int qg = q0 + wid * 16 + r + 8 * hf;
#pragma unroll
        for (int nd = 0; nd < 4; ++nd) {
            const size_t idx = ((size_t)(b * SEQ + qg)) * D_MODEL + h * HDIM + nd * 16 + nl;
            O[idx] = f2bf(o[nd][r] * inv);
        }
    }
}

// ---------------------------------------------------------------------------
// 4) Output projection: out = Ob @ Wo^T  (bf16 x bf16 -> f32)
//    Fully DMA-fed, double-buffered, one barrier per iteration.
// ---------------------------------------------------------------------------
__global__ __launch_bounds__(256) void k_outproj(const __bf16* __restrict__ X,
                                                 const __bf16* __restrict__ W,
                                                 float* __restrict__ out) {
    __shared__ __bf16 Xs[2][128 * 40];
    __shared__ __bf16 Ws[2][64 * 40];

    const int t = threadIdx.x;
    const int lane = t & 31, wid = t >> 5;
    const int wm = wid & 3, wn = wid >> 2;
    const int M0 = blockIdx.y * 128;
    const int N0 = blockIdx.x * 64;

    const int xrow = t >> 1;               // 0..127
    const int xcb  = (t & 1) * 16;         // 0,16
    const int wrow = t >> 2;               // 0..63
    const int wcc  = (t & 3) * 8;          // 0,8,16,24

    v8f acc[4] = {};

    // ---- prologue: stage tile k0=0 into buffer 0 ----
    {
        const __bf16* sx = X + (size_t)(M0 + xrow) * D_MODEL + xcb;
        const unsigned xd = lds_addr(&Xs[0][xrow * 40 + xcb]);
        async_copy_b128(xd,      sx);
        async_copy_b128(xd + 16, sx + 8);
        async_copy_b128(lds_addr(&Ws[0][wrow * 40 + wcc]),
                        W + (size_t)(N0 + wrow) * D_MODEL + wcc);
    }

    for (int k0 = 0; k0 < D_MODEL; k0 += 32) {
        const int cur = (k0 >> 5) & 1;
        const bool more = (k0 + 32) < D_MODEL;

        wait_async0();
        __syncthreads();

        if (more) {
            const int nxt = cur ^ 1;
            const __bf16* sx = X + (size_t)(M0 + xrow) * D_MODEL + (k0 + 32) + xcb;
            const unsigned xd = lds_addr(&Xs[nxt][xrow * 40 + xcb]);
            async_copy_b128(xd,      sx);
            async_copy_b128(xd + 16, sx + 8);
            async_copy_b128(lds_addr(&Ws[nxt][wrow * 40 + wcc]),
                            W + (size_t)(N0 + wrow) * D_MODEL + (k0 + 32) + wcc);
        }

        v16bf a0 = load_frag(Xs[cur], wm * 32,      40, lane);
        v16bf a1 = load_frag(Xs[cur], wm * 32 + 16, 40, lane);
        v16bf b0 = load_frag(Ws[cur], wn * 32,      40, lane);
        v16bf b1 = load_frag(Ws[cur], wn * 32 + 16, 40, lane);
        acc[0] = __builtin_amdgcn_wmma_f32_16x16x32_bf16(false, a0, false, b0, (short)0, acc[0], false, false);
        acc[1] = __builtin_amdgcn_wmma_f32_16x16x32_bf16(false, a0, false, b1, (short)0, acc[1], false, false);
        acc[2] = __builtin_amdgcn_wmma_f32_16x16x32_bf16(false, a1, false, b0, (short)0, acc[2], false, false);
        acc[3] = __builtin_amdgcn_wmma_f32_16x16x32_bf16(false, a1, false, b1, (short)0, acc[3], false, false);
    }

    const int hf = lane >> 4;
    const int nl = lane & 15;
#pragma unroll
    for (int mi = 0; mi < 2; ++mi)
#pragma unroll
    for (int ni = 0; ni < 2; ++ni)
#pragma unroll
    for (int r = 0; r < 8; ++r) {
        const int g = M0 + wm * 32 + mi * 16 + r + 8 * hf;
        const int c = N0 + wn * 32 + ni * 16 + nl;
        out[(size_t)g * D_MODEL + c] = acc[mi * 2 + ni][r];
    }
}

// ---------------------------------------------------------------------------
// launch
// ---------------------------------------------------------------------------
extern "C" void kernel_launch(void* const* d_in, const int* in_sizes, int n_in,
                              void* d_out, int out_size, void* d_ws, size_t ws_size,
                              hipStream_t stream) {
    const float* q  = (const float*)d_in[0];
    const float* k  = (const float*)d_in[1];
    const float* v  = (const float*)d_in[2];
    // d_in[3] = causal mask (handled analytically)
    const float* Wq = (const float*)d_in[4];
    const float* Wk = (const float*)d_in[5];
    const float* Wv = (const float*)d_in[6];
    const float* Wo = (const float*)d_in[7];

    __bf16* ws = (__bf16*)d_ws;
    const size_t DD  = (size_t)D_MODEL * D_MODEL;                 // 1 Mi elems
    const size_t QKV = (size_t)BATCH * NHEADS * SEQ * HDIM;       // 8 Mi elems
    __bf16* Wb = ws;               // 4*DD   (Wq,Wk,Wv,Wo bf16)
    __bf16* Qb = ws + 4 * DD;      // [B,H,S,Dh]
    __bf16* Kb = Qb + QKV;         // [B,H,S,Dh]
    __bf16* Vb = Kb + QKV;         // [B,H,Dh,S]  (transposed)
    __bf16* Ob = Vb + QKV;         // [B,S,D]
    (void)in_sizes; (void)n_in; (void)out_size; (void)ws_size;

    const int cgrd = (int)(DD / 4 / 256);
    k_f32_to_bf16<<<cgrd, 256, 0, stream>>>(Wq, Wb + 0 * DD, (int)DD);
    k_f32_to_bf16<<<cgrd, 256, 0, stream>>>(Wk, Wb + 1 * DD, (int)DD);
    k_f32_to_bf16<<<cgrd, 256, 0, stream>>>(Wv, Wb + 2 * DD, (int)DD);
    k_f32_to_bf16<<<cgrd, 256, 0, stream>>>(Wo, Wb + 3 * DD, (int)DD);

    k_proj<<<dim3(D_MODEL / 64, (BATCH * SEQ) / 128, 3), 256, 0, stream>>>(
        q, k, v, Wb, Qb, Kb, Vb);

    k_flash<<<dim3(SEQ / 64, BATCH * NHEADS), 128, 0, stream>>>(Qb, Kb, Vb, Ob);

    k_outproj<<<dim3(D_MODEL / 64, (BATCH * SEQ) / 128), 256, 0, stream>>>(
        Ob, Wb + 3 * DD, (float*)d_out);
}
